// Simulator_80487687127343
// MI455X (gfx1250) — matrix operand order repaired and checked by reading, not compiled, not run on hardware
//
#include <hip/hip_runtime.h>

#define NNODES 10000
#define NEDGES 320000

typedef __attribute__((ext_vector_type(16))) _Float16 v16h;
typedef __attribute__((ext_vector_type(8)))  _Float16 v8h;
typedef __attribute__((ext_vector_type(8)))  float    v8f;
typedef __attribute__((ext_vector_type(4)))  unsigned u32x4;
typedef __attribute__((ext_vector_type(8)))  int      i32x8;
typedef __attribute__((ext_vector_type(4)))  int      i32x4;

union V16 { v16h v; v8h h[2]; };
union V8F { v8f v; float f[8]; };

__device__ __forceinline__ int padw(int w) { return (w && w < 32) ? 32 : w; }

// Flat LDS pointer -> wave-relative LDS byte offset (flat addr[31:0] == LDS offset).
__device__ __forceinline__ unsigned lds_off(const void* p)
{
    return (unsigned)(unsigned long long)p;
}

// Async DMA 16B global -> LDS, tracked by ASYNCcnt (GV mode, saddr = off).
__device__ __forceinline__ void async_copy16(const void* gsrc, void* ldst)
{
    unsigned d = lds_off(ldst);
    unsigned long long a = (unsigned long long)gsrc;
    asm volatile("global_load_async_to_lds_b128 %0, %1, off"
                 :: "v"(d), "v"(a) : "memory");
}
__device__ __forceinline__ void wait_async()
{
    asm volatile("s_wait_asynccnt 0x0" ::: "memory");
}

// TDM: 1D DMA of `bytes` (multiple of 8) global -> LDS via tensor_load_to_lds.
// D# group0: count=1, lds_addr, global_addr[56:0], type=2.
// D# group1: data_size=3 (8B), tensor_dim0=tile_dim0=elems, dim1=1.
__device__ __forceinline__ void tdm_load_1d(const void* gsrc, void* ldst, int bytes)
{
    unsigned long long ga = (unsigned long long)gsrc;
    unsigned lds = lds_off(ldst);
    int elems = bytes >> 3;
    u32x4 g0;
    g0.x = 1u;                                                    // count=1 (valid)
    g0.y = lds;                                                   // lds_addr
    g0.z = (unsigned)ga;                                          // global_addr lo
    g0.w = (unsigned)((ga >> 32) & 0x1FFFFFFull) | (2u << 30);    // ga[56:32] | type=2
    i32x8 g1;
    g1[0] = 0x00030000;                           // data_size=3 (8 bytes)
    g1[1] = (elems & 0xffff) << 16;               // tensor_dim0[15:0]
    g1[2] = ((elems >> 16) & 0xffff) | (1 << 16); // tensor_dim0[31:16], tensor_dim1=1
    g1[3] = (elems & 0xffff) << 16;               // tile_dim0 = elems
    g1[4] = 0;                                    // tile_dim1=0 (1D), tile_dim2=0
    g1[5] = elems;                                // tensor_dim0_stride[31:0]
    g1[6] = 0;
    g1[7] = 0;
    i32x4 z4 = { 0, 0, 0, 0 };
    i32x8 z8 = { 0, 0, 0, 0, 0, 0, 0, 0 };
    __builtin_amdgcn_tensor_load_to_lds(g0, g1, z4, z4, z8, 0);
}

// ---------------------------------------------------------------------------
// Weight swizzle: f32 [K,128] row-major -> f16 WMMA-B fragment-ready order.
// Per 32-K chunk: [tile t=n/16][lane = n%16 + 16*(kk/16)][j = kk%16]
// ---------------------------------------------------------------------------
__global__ void k_swizzle_w(const float* __restrict__ src, _Float16* __restrict__ dst,
                            int K, int padK)
{
    int i = blockIdx.x * 256 + threadIdx.x;
    int total = padK * 128;
    if (i >= total) return;
    int k = i >> 7, n = i & 127;
    float val = (k < K) ? src[k * 128 + n] : 0.f;
    int ch = k >> 5, kk = k & 31;
    int t = n >> 4, nn = n & 15;
    int lane = nn + ((kk >> 4) << 4);
    int j = kk & 15;
    dst[(size_t)ch * 4096 + t * 512 + lane * 16 + j] = (_Float16)val;
}

__global__ void k_f32_to_f16(const float* __restrict__ s, _Float16* __restrict__ d, int n)
{
    int i = blockIdx.x * 256 + threadIdx.x;
    if (i < n) d[i] = (_Float16)s[i];
}

__global__ void k_zero_f32(float* __restrict__ d, int n)
{
    int i = blockIdx.x * 256 + threadIdx.x;
    if (i < n) d[i] = 0.f;
}

__global__ void k_edge_feat(const float* __restrict__ pos, const int* __restrict__ snd,
                            const int* __restrict__ rcv, _Float16* __restrict__ ef, int E)
{
    int e = blockIdx.x * 256 + threadIdx.x;
    if (e >= E) return;
    int a = snd[e], b = rcv[e];
    const float inv_r = 1.f / 0.03f;
    float rx = (pos[b * 3 + 0] - pos[a * 3 + 0]) * inv_r;
    float ry = (pos[b * 3 + 1] - pos[a * 3 + 1]) * inv_r;
    float rz = (pos[b * 3 + 2] - pos[a * 3 + 2]) * inv_r;
    float d = sqrtf(rx * rx + ry * ry + rz * rz);
    ef[(size_t)e * 4 + 0] = (_Float16)rx;
    ef[(size_t)e * 4 + 1] = (_Float16)ry;
    ef[(size_t)e * 4 + 2] = (_Float16)rz;
    ef[(size_t)e * 4 + 3] = (_Float16)d;
}

__global__ void k_scatter(const _Float16* __restrict__ e16, const int* __restrict__ rcv,
                          float* __restrict__ agg, int E)
{
    int i = blockIdx.x * 256 + threadIdx.x;
    if (i >= E * 32) return;
    int e = i >> 5;
    int g = (i & 31) << 2;
    int r = rcv[e];
    const _Float16* ep = e16 + (size_t)e * 128 + g;
    float* ap = agg + (size_t)r * 128 + g;
#pragma unroll
    for (int j = 0; j < 4; j++) atomicAdd(ap + j, (float)ep[j]);
}

// ---------------------------------------------------------------------------
// Fused GEMM: out[M,128] = epilogue( concat-A[M,K] @ Wsw[K,128] + bias )
// Block = 256 thr = 8 waves; wave owns 32 rows (2 WMMA M-tiles) x 128 cols.
// B chunks: TDM tensor_load_to_lds (wave 0), double-buffered.
// A tiles : global_load_async_to_lds_b128 per lane, double-buffered.
// Epilogue: ReLU, or LayerNorm (+optional f16 residual) via 16-lane shfl_xor.
// ---------------------------------------------------------------------------
__global__ __launch_bounds__(256)
void k_gemm128(const _Float16* __restrict__ a0, const int* __restrict__ i0, int w0,
               const _Float16* __restrict__ a1, const int* __restrict__ i1, int w1,
               const _Float16* __restrict__ a2, const int* __restrict__ i2, int w2,
               int M,
               const _Float16* __restrict__ Wsw, const float* __restrict__ bias,
               const float* __restrict__ lng, const float* __restrict__ lnb,
               const _Float16* __restrict__ resid,
               _Float16* __restrict__ out)
{
    __shared__ _Float16 ldsB[2][4096];      // 2 x 8 KB swizzled weight chunk
    __shared__ _Float16 ldsA[2][8][1024];   // 2 x (per-wave 32 rows x 32 K)

    const int tid  = threadIdx.x;
    const int w    = tid >> 5;
    const int lane = tid & 31;
    const int m    = lane & 15;
    const int c    = lane >> 4;
    const int nn   = m;

    const int p0 = padw(w0), p1 = padw(w1), p2 = padw(w2);
    const int K  = p0 + p1 + p2;
    const int nch = K >> 5;

    const int rowBase = blockIdx.x * 256 + w * 32;
    int gr0 = rowBase + m;      if (gr0 > M - 1) gr0 = M - 1;
    int gr1 = rowBase + 16 + m; if (gr1 > M - 1) gr1 = M - 1;

    V8F zero;
#pragma unroll
    for (int j = 0; j < 8; j++) zero.f[j] = 0.f;
    v8f acc0[8], acc1[8];
#pragma unroll
    for (int t = 0; t < 8; t++) { acc0[t] = zero.v; acc1[t] = zero.v; }

    auto stageB = [&](int ch, int buf) {
        if (w == 0)
            tdm_load_1d(Wsw + (size_t)ch * 4096, &ldsB[buf][0], 8192);
    };

    auto stageA = [&](int ch, int buf) {
        int k0 = ch << 5;
        const _Float16* sp; const int* si; int sw; int koff;
        if (k0 < p0)           { sp = a0; si = i0; sw = w0; koff = k0; }
        else if (k0 < p0 + p1) { sp = a1; si = i1; sw = w1; koff = k0 - p0; }
        else                   { sp = a2; si = i2; sw = w2; koff = k0 - p0 - p1; }
        int r0 = si ? si[gr0] : gr0;
        int r1 = si ? si[gr1] : gr1;
        int kk0 = koff + c * 16;
        _Float16* d0 = &ldsA[buf][w][m * 32 + c * 16];
        _Float16* d1 = &ldsA[buf][w][(m + 16) * 32 + c * 16];
        if (sw == 128) {
            const _Float16* s0 = sp + (size_t)r0 * 128 + kk0;
            const _Float16* s1 = sp + (size_t)r1 * 128 + kk0;
            async_copy16(s0,     d0);
            async_copy16(s0 + 8, d0 + 8);
            async_copy16(s1,     d1);
            async_copy16(s1 + 8, d1 + 8);
        } else {                                  // encoder first layer (width 4)
#pragma unroll
            for (int j = 0; j < 16; j++) {
                int kk = kk0 + j;
                d0[j] = (kk < sw) ? sp[(size_t)r0 * sw + kk] : (_Float16)0.f;
                d1[j] = (kk < sw) ? sp[(size_t)r1 * sw + kk] : (_Float16)0.f;
            }
        }
    };

    auto computeChunk = [&](int buf) {
        V16 af0, af1;
        const _Float16* ap = &ldsA[buf][w][0];
        af0.h[0] = *(const v8h*)(ap + m * 32 +      c * 16);
        af0.h[1] = *(const v8h*)(ap + m * 32 + 8 + c * 16);
        af1.h[0] = *(const v8h*)(ap + (m + 16) * 32 +      c * 16);
        af1.h[1] = *(const v8h*)(ap + (m + 16) * 32 + 8 + c * 16);
        const _Float16* bbase = &ldsB[buf][0];
        V16 bf[2];
        bf[0].h[0] = *(const v8h*)(bbase + lane * 16);
        bf[0].h[1] = *(const v8h*)(bbase + lane * 16 + 8);
#pragma unroll
        for (int t = 0; t < 8; t++) {
            if (t < 7) {                           // software pipeline next B frag
                const _Float16* bp = bbase + (t + 1) * 512 + lane * 16;
                bf[(t + 1) & 1].h[0] = *(const v8h*)(bp);
                bf[(t + 1) & 1].h[1] = *(const v8h*)(bp + 8);
            }
            acc0[t] = __builtin_amdgcn_wmma_f32_16x16x32_f16(
                false, af0.v, false, bf[t & 1].v, (short)0, acc0[t], false, false);
            acc1[t] = __builtin_amdgcn_wmma_f32_16x16x32_f16(
                false, af1.v, false, bf[t & 1].v, (short)0, acc1[t], false, false);
        }
    };

    // prologue: stage chunk 0 into buffer 0
    stageB(0, 0);
    stageA(0, 0);
    if (w == 0) __builtin_amdgcn_s_wait_tensorcnt(0);
    wait_async();
    __syncthreads();

    for (int ch = 0; ch < nch; ++ch) {
        int cur = ch & 1, nxt = cur ^ 1;
        if (ch + 1 < nch) {                        // overlap DMA of next chunk
            stageB(ch + 1, nxt);
            stageA(ch + 1, nxt);
        }
        computeChunk(cur);
        if (w == 0) __builtin_amdgcn_s_wait_tensorcnt(0);
        wait_async();
        __syncthreads();
    }

    // ---- epilogue ----
    float bn[8], gg[8], bb[8];
#pragma unroll
    for (int t = 0; t < 8; t++) {
        bn[t] = bias[t * 16 + nn];
        gg[t] = lng ? lng[t * 16 + nn] : 0.f;
        bb[t] = lnb ? lnb[t * 16 + nn] : 0.f;
    }

#pragma unroll
    for (int mt = 0; mt < 2; ++mt) {
        V8F ac[8];
#pragma unroll
        for (int t = 0; t < 8; t++) ac[t].v = mt ? acc1[t] : acc0[t];
        const int rb = rowBase + mt * 16;

        if (!lng) {
#pragma unroll
            for (int r = 0; r < 8; r++) {
                int row = rb + r + 8 * c;          // VGPR r holds rows r / r+8
                bool ok = row < M;
#pragma unroll
                for (int t = 0; t < 8; t++) {
                    float v = ac[t].f[r] + bn[t];
                    v = v > 0.f ? v : 0.f;
                    if (ok) out[(size_t)row * 128 + t * 16 + nn] = (_Float16)v;
                }
            }
        } else {
#pragma unroll
            for (int r = 0; r < 8; r++) {
                float s = 0.f, q = 0.f;
#pragma unroll
                for (int t = 0; t < 8; t++) {
                    float v = ac[t].f[r] + bn[t];
                    ac[t].f[r] = v;
                    s += v; q += v * v;
                }
#pragma unroll
                for (int msk = 1; msk <= 8; msk <<= 1) {
                    s += __shfl_xor(s, msk, 32);
                    q += __shfl_xor(q, msk, 32);
                }
                float mu  = s * (1.f / 128.f);
                float var = q * (1.f / 128.f) - mu * mu;
                float rs  = rsqrtf(var + 1e-5f);
                int row = rb + r + 8 * c;
                bool ok = row < M;
#pragma unroll
                for (int t = 0; t < 8; t++) {
                    float v = (ac[t].f[r] - mu) * rs * gg[t] + bb[t];
                    if (ok) {
                        if (resid) v += (float)resid[(size_t)row * 128 + t * 16 + nn];
                        out[(size_t)row * 128 + t * 16 + nn] = (_Float16)v;
                    }
                }
            }
        }
    }
}

__global__ void k_dec_final(const _Float16* __restrict__ h, const float* __restrict__ W,
                            const float* __restrict__ b, const float* __restrict__ x,
                            float* __restrict__ out, int N)
{
    int row = blockIdx.x * 256 + threadIdx.x;
    if (row >= N) return;
    float a0 = b[0], a1 = b[1], a2 = b[2], a3 = b[3];
    const _Float16* hp = h + (size_t)row * 128;
    for (int k = 0; k < 128; k++) {
        float hv = (float)hp[k];
        a0 += hv * W[k * 4 + 0];
        a1 += hv * W[k * 4 + 1];
        a2 += hv * W[k * 4 + 2];
        a3 += hv * W[k * 4 + 3];
    }
    out[row * 4 + 0] = a0;
    out[row * 4 + 1] = x[row * 4 + 1] + a1;
    out[row * 4 + 2] = x[row * 4 + 2] + a2;
    out[row * 4 + 3] = x[row * 4 + 3] + a3;
}

// ---------------------------------------------------------------------------
extern "C" void kernel_launch(void* const* d_in, const int* in_sizes, int n_in,
                              void* d_out, int out_size, void* d_ws, size_t ws_size,
                              hipStream_t stream)
{
    (void)in_sizes; (void)n_in; (void)out_size; (void)ws_size;
    const float* x   = (const float*)d_in[0];
    const float* pos = (const float*)d_in[1];
    const int*   snd = (const int*)d_in[2];
    const int*   rcv = snd + NEDGES;

    // Param leaf order (insertion order of setup_inputs dict):
    // 3..10 node_enc (W0 b0 W1 b1 W2 b2 g b), 11..18 edge_enc,
    // 19+16s step s: edge(+0..7) node(+8..15), 179..184 dec.
    auto P = [&](int i) { return (const float*)d_in[i]; };

    char* wp = (char*)d_ws;
    auto alloc = [&](size_t bytes) -> void* {
        void* r = wp; wp += (bytes + 255) & ~(size_t)255; return r;
    };

    _Float16* x16   = (_Float16*)alloc((size_t)NNODES * 4 * 2);
    _Float16* ef16  = (_Float16*)alloc((size_t)NEDGES * 4 * 2);
    _Float16* v16   = (_Float16*)alloc((size_t)NNODES * 128 * 2);
    _Float16* e16   = (_Float16*)alloc((size_t)NEDGES * 128 * 2);
    _Float16* tE1   = (_Float16*)alloc((size_t)NEDGES * 128 * 2);
    _Float16* tE2   = (_Float16*)alloc((size_t)NEDGES * 128 * 2);
    _Float16* tN1   = (_Float16*)alloc((size_t)NNODES * 128 * 2);
    _Float16* tN2   = (_Float16*)alloc((size_t)NNODES * 128 * 2);
    float*    agg   = (float*)   alloc((size_t)NNODES * 128 * 4);
    _Float16* agg16 = (_Float16*)alloc((size_t)NNODES * 128 * 2);

    auto swz = [&](int pidx, int K) -> _Float16* {
        int padK = (K < 32) ? 32 : K;
        _Float16* dst = (_Float16*)alloc((size_t)padK * 128 * 2);
        int total = padK * 128;
        k_swizzle_w<<<(total + 255) / 256, 256, 0, stream>>>(P(pidx), dst, K, padK);
        return dst;
    };

    _Float16* Wne[3] = { swz(3, 4),  swz(5, 128),  swz(7, 128)  };
    _Float16* Wee[3] = { swz(11, 4), swz(13, 128), swz(15, 128) };
    _Float16* Wse[10][3]; _Float16* Wsn[10][3];
    for (int s = 0; s < 10; s++) {
        int be = 19 + 16 * s, bn = be + 8;
        Wse[s][0] = swz(be + 0, 384); Wse[s][1] = swz(be + 2, 128); Wse[s][2] = swz(be + 4, 128);
        Wsn[s][0] = swz(bn + 0, 256); Wsn[s][1] = swz(bn + 2, 128); Wsn[s][2] = swz(bn + 4, 128);
    }
    _Float16* Wd[2] = { swz(179, 128), swz(181, 128) };

    auto gemm = [&](const _Float16* a0, const int* i0, int w0,
                    const _Float16* a1, const int* i1, int w1,
                    const _Float16* a2, const int* i2, int w2,
                    int M, const _Float16* Wsw, const float* bias,
                    const float* g, const float* b, const _Float16* resid,
                    _Float16* out) {
        int blocks = (M + 255) / 256;
        k_gemm128<<<blocks, 256, 0, stream>>>(a0, i0, w0, a1, i1, w1, a2, i2, w2,
                                              M, Wsw, bias, g, b, resid, out);
    };
    const _Float16* NH = nullptr; const int* NI = nullptr;

    k_f32_to_f16<<<(NNODES * 4 + 255) / 256, 256, 0, stream>>>(x, x16, NNODES * 4);
    k_edge_feat<<<(NEDGES + 255) / 256, 256, 0, stream>>>(pos, snd, rcv, ef16, NEDGES);

    // encoders: relu, relu, LN (no residual)
    gemm(x16, NI, 4,   NH, NI, 0, NH, NI, 0, NNODES, Wne[0], P(4),  nullptr, nullptr, NH, tN1);
    gemm(tN1, NI, 128, NH, NI, 0, NH, NI, 0, NNODES, Wne[1], P(6),  nullptr, nullptr, NH, tN2);
    gemm(tN2, NI, 128, NH, NI, 0, NH, NI, 0, NNODES, Wne[2], P(8),  P(9),  P(10), NH, v16);

    gemm(ef16, NI, 4,   NH, NI, 0, NH, NI, 0, NEDGES, Wee[0], P(12), nullptr, nullptr, NH, tE1);
    gemm(tE1,  NI, 128, NH, NI, 0, NH, NI, 0, NEDGES, Wee[1], P(14), nullptr, nullptr, NH, tE2);
    gemm(tE2,  NI, 128, NH, NI, 0, NH, NI, 0, NEDGES, Wee[2], P(16), P(17), P(18), NH, e16);

    // message-passing steps
    for (int s = 0; s < 10; s++) {
        int be = 19 + 16 * s, bn = be + 8;
        gemm(e16, NI, 128, v16, snd, 128, v16, rcv, 128,
             NEDGES, Wse[s][0], P(be + 1), nullptr, nullptr, NH, tE1);
        gemm(tE1, NI, 128, NH, NI, 0, NH, NI, 0,
             NEDGES, Wse[s][1], P(be + 3), nullptr, nullptr, NH, tE2);
        gemm(tE2, NI, 128, NH, NI, 0, NH, NI, 0,
             NEDGES, Wse[s][2], P(be + 5), P(be + 6), P(be + 7), e16, e16);

        k_zero_f32<<<(NNODES * 128 + 255) / 256, 256, 0, stream>>>(agg, NNODES * 128);
        k_scatter<<<(NEDGES * 32 + 255) / 256, 256, 0, stream>>>(e16, rcv, agg, NEDGES);
        k_f32_to_f16<<<(NNODES * 128 + 255) / 256, 256, 0, stream>>>(agg, agg16, NNODES * 128);

        gemm(v16, NI, 128, agg16, NI, 128, NH, NI, 0,
             NNODES, Wsn[s][0], P(bn + 1), nullptr, nullptr, NH, tN1);
        gemm(tN1, NI, 128, NH, NI, 0, NH, NI, 0,
             NNODES, Wsn[s][1], P(bn + 3), nullptr, nullptr, NH, tN2);
        gemm(tN2, NI, 128, NH, NI, 0, NH, NI, 0,
             NNODES, Wsn[s][2], P(bn + 5), P(bn + 6), P(bn + 7), v16, v16);
    }

    // decoder
    gemm(v16, NI, 128, NH, NI, 0, NH, NI, 0, NNODES, Wd[0], P(180), nullptr, nullptr, NH, tN1);
    gemm(tN1, NI, 128, NH, NI, 0, NH, NI, 0, NNODES, Wd[1], P(182), nullptr, nullptr, NH, tN2);
    k_dec_final<<<(NNODES + 255) / 256, 256, 0, stream>>>(tN2, P(183), P(184), x,
                                                          (float*)d_out, NNODES);
}
